// OrthogonalTransform_70746701300434
// MI455X (gfx1250) — matrix operand order repaired and checked by reading, not compiled, not run on hardware
//
#include <hip/hip_runtime.h>

typedef __attribute__((ext_vector_type(16))) __bf16 v16bf;
typedef __attribute__((ext_vector_type(8)))  float  v8f;

#define RDIM   128
#define KPAD   136   // padded K stride (halfs): 272B row stride -> conflict-free
                     // ds_load_b128 across 16 lanes, 16B aligned
#define MTILES 4     // 16-row m-tiles per wave; block covers 512 rows

union AFrag { v16bf v; unsigned u[8]; };
union BFrag { v16bf v; uint4 q[2]; };

// Pack top 16 bits of two fp32 dwords into one dword: {e0>>16 | (e1>>16)<<16}.
// Single v_perm_b32: byte sel 0..3 -> src1 (e0), 4..7 -> src0 (e1).
static __device__ __forceinline__ unsigned pack_hi16(unsigned e1, unsigned e0) {
    return __builtin_amdgcn_perm(e1, e0, 0x07060302u);
}

__global__ __launch_bounds__(256)
void OrthogonalTransform_rot128_wmma_bf16x3(const float* __restrict__ x,
                                            const float* __restrict__ R,
                                            float* __restrict__ y,
                                            int M) {
    __shared__ unsigned short sHiT[RDIM * KPAD];   // R hi-plane (trunc), transposed [n][k]
    __shared__ unsigned short sLoT[RDIM * KPAD];   // R lo-plane (residual), transposed [n][k]

    const int tid = threadIdx.x;

    // Cooperative fill, 32 iterations: each thread handles a (k, k+1) pair for one n.
    // Dekker split: hi = bits & 0xFFFF0000 (exact as float), lo = f - hi (exact),
    // both planes stored as packed bf16 pairs with one v_perm each.
    for (int p = tid; p < 64 * RDIM; p += 256) {
        const int n = p & 127;
        const int k = (p >> 7) << 1;
        const unsigned u0 = __float_as_uint(R[(size_t)k * RDIM + n]);
        const unsigned u1 = __float_as_uint(R[(size_t)(k + 1) * RDIM + n]);
        const unsigned h0 = u0 & 0xFFFF0000u;
        const unsigned h1 = u1 & 0xFFFF0000u;
        const float l0 = __uint_as_float(u0) - __uint_as_float(h0);
        const float l1 = __uint_as_float(u1) - __uint_as_float(h1);
        *(unsigned*)&sHiT[n * KPAD + k] = pack_hi16(u1, u0);
        *(unsigned*)&sLoT[n * KPAD + k] = pack_hi16(__float_as_uint(l1), __float_as_uint(l0));
    }
    __syncthreads();

    const int lane    = tid & 31;
    const int wave    = tid >> 5;
    const int hiHalf  = lane >> 4;                // 0: lanes 0-15, 1: lanes 16-31
    const int akoff   = hiHalf ? 8  : 0;          // A-layout K offset for this half-wave
    const int bkoff   = hiHalf ? 8  : 0;          // B-layout K offset for this half-wave
    const int mrow    = lane & 15;
    const int blockBase = blockIdx.x * (128 * MTILES);

    for (int mt = 0; mt < MTILES; ++mt) {
        const int rowBase = blockBase + mt * 128 + wave * 16;
        if (rowBase + 16 > M) continue;           // wave-uniform: EXEC stays all-1s

        const float* xrow = x + (size_t)(rowBase + mrow) * RDIM;

        v8f acc[8] = {};                          // 8 n-tiles of 16x16 fp32 accumulators

        #pragma unroll
        for (int ks = 0; ks < 4; ++ks) {          // K = 128 in steps of 32
            const int k0 = ks * 32;
            // A source: 16 fp32 per lane in the 16-bit A VGPR order
            // (lanes 0-15: K=k0+0..7, k0+16..23 ; lanes 16-31: K=k0+8..15, k0+24..31)
            const uint4 q0 = *(const uint4*)(xrow + k0 + akoff);
            const uint4 q1 = *(const uint4*)(xrow + k0 + akoff + 4);
            const uint4 q2 = *(const uint4*)(xrow + k0 + akoff + 16);
            const uint4 q3 = *(const uint4*)(xrow + k0 + akoff + 20);
            const unsigned u[16] = {q0.x,q0.y,q0.z,q0.w, q1.x,q1.y,q1.z,q1.w,
                                    q2.x,q2.y,q2.z,q2.w, q3.x,q3.y,q3.z,q3.w};
            AFrag ah, al;
            #pragma unroll
            for (int j = 0; j < 8; ++j) {
                const unsigned e0 = u[2*j], e1 = u[2*j+1];
                const unsigned h0 = e0 & 0xFFFF0000u;
                const unsigned h1 = e1 & 0xFFFF0000u;
                const float l0 = __uint_as_float(e0) - __uint_as_float(h0);
                const float l1 = __uint_as_float(e1) - __uint_as_float(h1);
                ah.u[j] = pack_hi16(e1, e0);
                al.u[j] = pack_hi16(__float_as_uint(l1), __float_as_uint(l0));
            }

            #pragma unroll
            for (int nt = 0; nt < 8; ++nt) {
                // B fragment: bf16 along K at bkoff+0..7, bkoff+16..23, column n = nt*16 + (lane&15)
                const int off = (nt * 16 + (lane & 15)) * KPAD + k0 + bkoff;
                BFrag bh, bl;
                bh.q[0] = *(const uint4*)&sHiT[off];
                bh.q[1] = *(const uint4*)&sHiT[off + 16];
                bl.q[0] = *(const uint4*)&sLoT[off];
                bl.q[1] = *(const uint4*)&sLoT[off + 16];

                // Split-precision accumulate: hi*hi + lo*hi + hi*lo (lo*lo dropped)
                acc[nt] = __builtin_amdgcn_wmma_f32_16x16x32_bf16(
                    false, ah.v, false, bh.v, (short)0, acc[nt], false, false);
                acc[nt] = __builtin_amdgcn_wmma_f32_16x16x32_bf16(
                    false, al.v, false, bh.v, (short)0, acc[nt], false, false);
                acc[nt] = __builtin_amdgcn_wmma_f32_16x16x32_bf16(
                    false, ah.v, false, bl.v, (short)0, acc[nt], false, false);
            }
        }

        // C/D layout: VGPR r -> M = r (lanes 0-15) / r+8 (lanes 16-31), N = lane&15
        const int moff = hiHalf ? 8 : 0;
        #pragma unroll
        for (int nt = 0; nt < 8; ++nt) {
            #pragma unroll
            for (int r = 0; r < 8; ++r) {
                y[(size_t)(rowBase + r + moff) * RDIM + nt * 16 + (lane & 15)] = acc[nt][r];
            }
        }
    }
}

extern "C" void kernel_launch(void* const* d_in, const int* in_sizes, int n_in,
                              void* d_out, int out_size, void* d_ws, size_t ws_size,
                              hipStream_t stream) {
    const float* x = (const float*)d_in[0];
    const float* R = (const float*)d_in[1];
    float* y = (float*)d_out;

    const long long total = (long long)in_sizes[0];     // 4*4096*4096
    const int M = (int)(total / RDIM);                  // 524288 rows of 128
    const int rowsPerBlock = 128 * MTILES;              // 512
    const int blocks = (M + rowsPerBlock - 1) / rowsPerBlock;   // 1024

    hipLaunchKernelGGL(OrthogonalTransform_rot128_wmma_bf16x3,
                       dim3(blocks), dim3(256), 0, stream, x, R, y, M);
}
